// EncoderLayer_83880711291016
// MI455X (gfx1250) — matrix operand order repaired and checked by reading, not compiled, not run on hardware
//
#include <hip/hip_runtime.h>
#include <stdint.h>

typedef __bf16 bf16;
typedef bf16  v8bf  __attribute__((ext_vector_type(8)));
typedef bf16  v16bf __attribute__((ext_vector_type(16)));
typedef float v8f   __attribute__((ext_vector_type(8)));

__device__ __forceinline__ bf16 f2bf(float f) {
  unsigned u = __builtin_bit_cast(unsigned, f);
  unsigned r = (u + 0x7FFFu + ((u >> 16) & 1u)) >> 16;   // round-to-nearest-even
  return __builtin_bit_cast(bf16, (unsigned short)r);
}

#define SHUF16(lo, hi) __builtin_shufflevector(lo, hi, 0,1,2,3,4,5,6,7,8,9,10,11,12,13,14,15)
#define WMMA_BF16(a, b, c) __builtin_amdgcn_wmma_f32_16x16x32_bf16( \
    false, a, false, b, (short)0, c, false, false)

// ---------------------------------------------------------------------------
// Elementwise f32 -> bf16 convert (row-major copy, used for A matrices)
// ---------------------------------------------------------------------------
__global__ void k_cvt(const float* __restrict__ in, bf16* __restrict__ out, int n) {
  int i = blockIdx.x * blockDim.x + threadIdx.x;
  if (i < n) out[i] = f2bf(in[i]);
}

// Transpose convert: in f32 [K,N] -> out bf16 [N,K], K = 1<<kshift (weights)
__global__ void k_cvt_T(const float* __restrict__ in, bf16* __restrict__ out,
                        int kshift, int N) {
  int i = blockIdx.x * blockDim.x + threadIdx.x;   // over N*K, coalesced writes
  int k = i & ((1 << kshift) - 1);
  int n = i >> kshift;
  out[i] = f2bf(in[(size_t)k * N + n]);
}

// [H=16, D=1024, dk=64] -> transposed combined [H*dk=1024, D=1024] bf16
__global__ void k_cvt_qkvw(const float* __restrict__ in, bf16* __restrict__ out) {
  int i = blockIdx.x * blockDim.x + threadIdx.x;   // H*D*dk threads, exact
  int d = i & 1023;          // column (k-dim of GEMM)
  int n = i >> 10;           // row  = h*64 + e
  int h = n >> 6;
  int e = n & 63;
  out[i] = f2bf(in[h * 65536 + d * 64 + e]);       // in[h][d][e]
}

// ---------------------------------------------------------------------------
// bf16 WMMA GEMM helpers
// ---------------------------------------------------------------------------
__device__ __forceinline__ void load_frags(
    const bf16* __restrict__ aRow0, const bf16* __restrict__ aRow1,
    const bf16* const* bCol, int k0,
    v16bf& a0, v16bf& a1, v16bf* bfr)
{
  a0 = SHUF16(*(const v8bf*)(aRow0 + k0), *(const v8bf*)(aRow0 + k0 + 16));
  a1 = SHUF16(*(const v8bf*)(aRow1 + k0), *(const v8bf*)(aRow1 + k0 + 16));
  #pragma unroll
  for (int et = 0; et < 4; ++et) bfr[et] = SHUF16(*(const v8bf*)(bCol[et] + k0), *(const v8bf*)(bCol[et] + k0 + 16));
}

__device__ __forceinline__ void mma_burst(
    const v16bf& a0, const v16bf& a1, const v16bf* bfr, v8f acc[2][4])
{
  #pragma unroll
  for (int et = 0; et < 4; ++et) {
    acc[0][et] = WMMA_BF16(a0, bfr[et], acc[0][et]);
    acc[1][et] = WMMA_BF16(a1, bfr[et], acc[1][et]);
  }
}

// ---------------------------------------------------------------------------
// bf16 WMMA GEMM: C[M,N] = A[M,K] * Bt^T  with Bt stored [N,K] (pre-transposed).
// Block: 256 threads (8 waves). Wave tile 32x64 (2 M-subtiles x 4 N-subtiles).
// Ping-pong double-buffered fragments (explicit 2x unroll): loads for chunk
// i+1 issue before the WMMA burst of chunk i, with no register rotation copies.
// mode 0: outb = bf16(acc + bias)
// mode 1: outf = acc + bias + res
// mode 2: outb = bf16(relu(acc + bias))
// ---------------------------------------------------------------------------
__global__ __launch_bounds__(256) void k_gemm(
    const bf16* __restrict__ A, const bf16* __restrict__ Bt,
    const float* __restrict__ bias, const float* __restrict__ res,
    bf16* __restrict__ outb, float* __restrict__ outf,
    int M, int N, int K, int mode)
{
  const int lane  = threadIdx.x & 31;
  const int w     = threadIdx.x >> 5;
  const int khalf = lane >> 4;
  const int l16   = lane & 15;
  const int n0    = blockIdx.x * 64;
  const int m0    = blockIdx.y * 256 + w * 32;

  v8f acc[2][4];
  #pragma unroll
  for (int mt = 0; mt < 2; ++mt)
    #pragma unroll
    for (int et = 0; et < 4; ++et)
      #pragma unroll
      for (int j = 0; j < 8; ++j) acc[mt][et][j] = 0.f;

  const bf16* aRow0 = A + (size_t)(m0 + l16) * K + khalf * 8;
  const bf16* aRow1 = A + (size_t)(m0 + 16 + l16) * K + khalf * 8;
  const bf16* bCol[4];
  #pragma unroll
  for (int et = 0; et < 4; ++et)
    bCol[et] = Bt + (size_t)(n0 + et * 16 + l16) * K + khalf * 8;

  // ping-pong fragment buffers
  v16bf a0A, a1A, bA[4];
  v16bf a0B, a1B, bB[4];

  load_frags(aRow0, aRow1, bCol, 0, a0A, a1A, bA);

  int k0 = 32;
  for (; k0 + 32 < K; k0 += 64) {
    load_frags(aRow0, aRow1, bCol, k0, a0B, a1B, bB);
    __builtin_prefetch(aRow0 + k0 + 128, 0, 1);      // speculative stream-ahead
    __builtin_prefetch(aRow1 + k0 + 128, 0, 1);
    __builtin_prefetch(bCol[0] + k0 + 128, 0, 1);
    __builtin_prefetch(bCol[2] + k0 + 128, 0, 1);
    mma_burst(a0A, a1A, bA, acc);
    load_frags(aRow0, aRow1, bCol, k0 + 32, a0A, a1A, bA);
    mma_burst(a0B, a1B, bB, acc);
  }
  load_frags(aRow0, aRow1, bCol, K - 32, a0B, a1B, bB);
  mma_burst(a0A, a1A, bA, acc);
  mma_burst(a0B, a1B, bB, acc);

  // ---- writeback (C layout: lane col = l16, rows khalf*8 + r) ----
  #pragma unroll
  for (int mt = 0; mt < 2; ++mt) {
    #pragma unroll
    for (int et = 0; et < 4; ++et) {
      const int col = n0 + et * 16 + l16;
      const float bia = bias[col];
      #pragma unroll
      for (int r = 0; r < 8; ++r) {
        const size_t row = (size_t)(m0 + mt * 16 + khalf * 8 + r);
        float v = acc[mt][et][r] + bia;
        if (mode == 1) {
          outf[row * N + col] = v + res[row * N + col];
        } else {
          if (mode == 2) v = v > 0.f ? v : 0.f;
          outb[row * N + col] = f2bf(v);
        }
      }
    }
  }
}

// ---------------------------------------------------------------------------
// Flash attention: per block = one (b,h) and 64 query rows (4 waves x 16).
// Q,K,V,O are bf16 [B*S, 1024] with column = h*64 + e.  Keys in chunks of 32.
// ---------------------------------------------------------------------------
#define VT_STRIDE 40

__global__ __launch_bounds__(128) void k_attn(
    const bf16* __restrict__ Q, const bf16* __restrict__ Kb,
    const bf16* __restrict__ Vb, bf16* __restrict__ O)
{
  __shared__ bf16 sVT[64 * VT_STRIDE];   // V chunk transposed: [e 0..63][key 0..31]
  __shared__ bf16 sP[4 * 16 * 32];       // per-wave P tile (16 q x 32 k)

  const int S = 2048, DH = 1024;
  const int t     = threadIdx.x;
  const int lane  = t & 31;
  const int w     = t >> 5;
  const int khalf = lane >> 4;
  const int l16   = lane & 15;
  const int b     = blockIdx.y >> 4;
  const int h     = blockIdx.y & 15;
  const int qbase = blockIdx.x * 64 + w * 16;
  const size_t rowbase = (size_t)b * S;
  const int hcol  = h * 64;

  // Q fragments: 16x64 = two 16x32 A fragments
  v16bf aq[2];
  {
    const size_t qrow = rowbase + qbase + l16;
    #pragma unroll
    for (int f = 0; f < 2; ++f) {
      const bf16* p = Q + qrow * DH + hcol + f * 32 + khalf * 8;
      aq[f] = SHUF16(*(const v8bf*)p, *(const v8bf*)(p + 16));
    }
  }

  float m_i[8], l_i[8];
  v8f o[4];
  #pragma unroll
  for (int r = 0; r < 8; ++r) { m_i[r] = -1e30f; l_i[r] = 0.f; }
  #pragma unroll
  for (int et = 0; et < 4; ++et)
    #pragma unroll
    for (int j = 0; j < 8; ++j) o[et][j] = 0.f;

  bf16* myP = &sP[w * (16 * 32)];

  for (int c = 0; c < S / 32; ++c) {
    const int keybase = c * 32;

    // stage V^T (32 keys x 64 e): 128 threads x 16 elems
    {
      const int key = t >> 2;                  // 0..31
      const int eb  = (t & 3) * 16;            // 0,16,32,48
      const bf16* vp = Vb + (rowbase + keybase + key) * DH + hcol + eb;
      v16bf vv = *reinterpret_cast<const v16bf*>(vp);
      #pragma unroll
      for (int j = 0; j < 16; ++j) sVT[(eb + j) * VT_STRIDE + key] = vv[j];
    }
    __syncthreads();

    // scores S = Q K^T (two 16x16 tiles; B fragment reads contiguous K rows)
    v8f sc[2];
    #pragma unroll
    for (int nt = 0; nt < 2; ++nt) {
      const size_t krow = rowbase + keybase + nt * 16 + l16;
      v8f s;
      #pragma unroll
      for (int j = 0; j < 8; ++j) s[j] = 0.f;
      #pragma unroll
      for (int f = 0; f < 2; ++f) {
        const bf16* kp = Kb + krow * DH + hcol + f * 32 + khalf * 8;
        v16bf bk = SHUF16(*(const v8bf*)kp, *(const v8bf*)(kp + 16));
        s = WMMA_BF16(aq[f], bk, s);
      }
      sc[nt] = s;
    }

    // online softmax (rows khalf*8 + r live across lanes of the same half-wave)
    #pragma unroll
    for (int r = 0; r < 8; ++r) {
      float s0 = sc[0][r] * 0.125f;
      float s1 = sc[1][r] * 0.125f;
      float mx = fmaxf(s0, s1);
      #pragma unroll
      for (int msk = 1; msk < 16; msk <<= 1)
        mx = fmaxf(mx, __shfl_xor(mx, msk, 16));
      float mnew  = fmaxf(m_i[r], mx);
      float alpha = __expf(m_i[r] - mnew);
      float p0 = __expf(s0 - mnew);
      float p1 = __expf(s1 - mnew);
      float rs = p0 + p1;
      #pragma unroll
      for (int msk = 1; msk < 16; msk <<= 1)
        rs += __shfl_xor(rs, msk, 16);
      l_i[r] = l_i[r] * alpha + rs;
      m_i[r] = mnew;
      myP[(khalf * 8 + r) * 32 + l16]      = f2bf(p0);
      myP[(khalf * 8 + r) * 32 + 16 + l16] = f2bf(p1);
      #pragma unroll
      for (int et = 0; et < 4; ++et) o[et][r] *= alpha;
    }

    // reload P as an A fragment (16x32)
    const bf16* pp = myP + l16 * 32 + khalf * 8;
    v16bf ap = SHUF16(*(const v8bf*)pp, *(const v8bf*)(pp + 16));

    // O += P * V
    #pragma unroll
    for (int et = 0; et < 4; ++et) {
      const bf16* vp = &sVT[(et * 16 + l16) * VT_STRIDE + khalf * 8];
      v16bf bv = SHUF16(*(const v8bf*)vp, *(const v8bf*)(vp + 16));
      o[et] = WMMA_BF16(ap, bv, o[et]);
    }
    __syncthreads();
  }

  // finalize: divide by l and write bf16 attention output
  #pragma unroll
  for (int r = 0; r < 8; ++r) {
    const float inv = 1.0f / l_i[r];
    const size_t orow = rowbase + qbase + khalf * 8 + r;
    #pragma unroll
    for (int et = 0; et < 4; ++et)
      O[orow * DH + hcol + et * 16 + l16] = f2bf(o[et][r] * inv);
  }
}

// ---------------------------------------------------------------------------
// LayerNorm over D=1024, one row per block (256 threads x 4 elems).
// Writes f32 output and optionally a bf16 copy for the next GEMM.
// ---------------------------------------------------------------------------
__global__ __launch_bounds__(256) void k_ln(
    const float* __restrict__ y, const float* __restrict__ g,
    const float* __restrict__ bta, float* __restrict__ outf,
    bf16* __restrict__ outb)
{
  __shared__ float rs[256], rq[256];
  const int D = 1024;
  const int t = threadIdx.x;
  const size_t row = blockIdx.x;
  const float4 v = *reinterpret_cast<const float4*>(y + row * D + t * 4);
  rs[t] = v.x + v.y + v.z + v.w;
  rq[t] = v.x * v.x + v.y * v.y + v.z * v.z + v.w * v.w;
  __syncthreads();
  for (int off = 128; off > 0; off >>= 1) {
    if (t < off) { rs[t] += rs[t + off]; rq[t] += rq[t + off]; }
    __syncthreads();
  }
  const float mean = rs[0] * (1.f / D);
  const float var  = rq[0] * (1.f / D) - mean * mean;
  const float rstd = rsqrtf(var + 1e-5f);
  const float xv[4] = { v.x, v.y, v.z, v.w };
  #pragma unroll
  for (int j = 0; j < 4; ++j) {
    const int col = t * 4 + j;
    float ov = (xv[j] - mean) * rstd * g[col] + bta[col];
    outf[row * D + col] = ov;
    if (outb) outb[row * D + col] = f2bf(ov);
  }
}

// ---------------------------------------------------------------------------
extern "C" void kernel_launch(void* const* d_in, const int* in_sizes, int n_in,
                              void* d_out, int out_size, void* d_ws, size_t ws_size,
                              hipStream_t stream)
{
  const int B = 4, S = 2048, D = 1024, H = 16, dk = 64, F = 4096;
  const int BS = B * S;

  const float* x   = (const float*)d_in[0];
  const float* Wq  = (const float*)d_in[1];
  const float* bq  = (const float*)d_in[2];
  const float* Wk  = (const float*)d_in[3];
  const float* bk  = (const float*)d_in[4];
  const float* Wv  = (const float*)d_in[5];
  const float* bv  = (const float*)d_in[6];
  const float* Wo  = (const float*)d_in[7];
  const float* bo  = (const float*)d_in[8];
  const float* g1  = (const float*)d_in[9];
  const float* be1 = (const float*)d_in[10];
  const float* W1  = (const float*)d_in[11];
  const float* b1  = (const float*)d_in[12];
  const float* W2  = (const float*)d_in[13];
  const float* b2  = (const float*)d_in[14];
  const float* g2  = (const float*)d_in[15];
  const float* be2 = (const float*)d_in[16];

  // -------- workspace layout with lifetime-based aliasing --------
  char* ws = (char*)d_ws;
  size_t off = 0;
  auto alloc = [&](size_t bytes) -> char* {
    char* p = ws + off;
    off += (bytes + 255) & ~(size_t)255;
    return p;
  };
  const size_t szXB = (size_t)BS * D * 2;           // 16 MB
  char* regA  = alloc(4 * szXB);                    // XB,Q,K,V ; later H1 (BS*F*2 == 4*szXB)
  bf16* XB  = (bf16*)regA;
  bf16* Qb  = (bf16*)(regA + szXB);
  bf16* Kb  = (bf16*)(regA + 2 * szXB);
  bf16* Vb  = (bf16*)(regA + 3 * szXB);
  bf16* H1  = (bf16*)regA;                          // aliases XB/Q/K/V (dead by FFN1)
  bf16* WqT = (bf16*)alloc((size_t)D * D * 2);      // all weights stored [N,K] bf16
  bf16* WkT = (bf16*)alloc((size_t)D * D * 2);
  bf16* WvT = (bf16*)alloc((size_t)D * D * 2);
  bf16* WoT = (bf16*)alloc((size_t)D * D * 2);
  bf16* W1T = (bf16*)alloc((size_t)D * F * 2);
  bf16* W2T = (bf16*)alloc((size_t)F * D * 2);
  char* regC = alloc(szXB);                         // attention out ; later x1 bf16
  bf16* OB  = (bf16*)regC;
  bf16* x1b = (bf16*)regC;                          // aliases OB (dead after O-proj)
  char* regD = alloc((size_t)BS * D * 4);           // y1 ; later y2
  float* y1 = (float*)regD;
  float* y2 = (float*)regD;                         // aliases y1 (dead after LN1)
  float* x1f = (float*)alloc((size_t)BS * D * 4);
  if (ws_size < off) return;                        // workspace too small; bail

  const int T = 256;

  // -------- bf16 conversion / weight transpose pre-pass --------
  k_cvt<<<(BS * D) / T, T, 0, stream>>>(x, XB, BS * D);
  k_cvt_qkvw<<<(H * D * dk) / T, T, 0, stream>>>(Wq, WqT);
  k_cvt_qkvw<<<(H * D * dk) / T, T, 0, stream>>>(Wk, WkT);
  k_cvt_qkvw<<<(H * D * dk) / T, T, 0, stream>>>(Wv, WvT);
  k_cvt_T<<<(D * D) / T, T, 0, stream>>>(Wo, WoT, 10, D);   // K=1024
  k_cvt_T<<<(D * F) / T, T, 0, stream>>>(W1, W1T, 10, F);   // K=1024
  k_cvt_T<<<(F * D) / T, T, 0, stream>>>(W2, W2T, 12, D);   // K=4096

  // -------- QKV projections (bias; bf16 out) --------
  dim3 gD(D / 64, BS / 256);
  k_gemm<<<gD, 256, 0, stream>>>(XB, WqT, bq, nullptr, Qb, nullptr, BS, D, D, 0);
  k_gemm<<<gD, 256, 0, stream>>>(XB, WkT, bk, nullptr, Kb, nullptr, BS, D, D, 0);
  k_gemm<<<gD, 256, 0, stream>>>(XB, WvT, bv, nullptr, Vb, nullptr, BS, D, D, 0);

  // -------- flash attention --------
  dim3 ga(S / 64, B * H);
  k_attn<<<ga, 128, 0, stream>>>(Qb, Kb, Vb, OB);

  // -------- output projection + residual, LN1 --------
  k_gemm<<<gD, 256, 0, stream>>>(OB, WoT, bo, x, nullptr, y1, BS, D, D, 1);
  k_ln<<<BS, 256, 0, stream>>>(y1, g1, be1, x1f, x1b);

  // -------- FFN --------
  dim3 gF(F / 64, BS / 256);
  k_gemm<<<gF, 256, 0, stream>>>(x1b, W1T, b1, nullptr, H1, nullptr, BS, F, D, 2);
  k_gemm<<<gD, 256, 0, stream>>>(H1, W2T, b2, x1f, nullptr, y2, BS, D, F, 1);

  // -------- LN2 -> final output --------
  k_ln<<<BS, 256, 0, stream>>>(y2, g2, be2, (float*)d_out, nullptr);
}
